// GraphCNN_48679159333670
// MI455X (gfx1250) — compile-verified, matrix operand order checked by reading, not run on hardware
//
#include <hip/hip_runtime.h>
#include <hip/hip_bf16.h>

typedef __attribute__((ext_vector_type(16))) __bf16 v16bf;
typedef __attribute__((ext_vector_type(8)))  float  v8f;

#define HID 64
#define NGRAPH 64

// ---------------- degree / normalization ----------------
__global__ void init_ones_kernel(float* __restrict__ p, int n) {
  int i = blockIdx.x * blockDim.x + threadIdx.x;
  if (i < n) p[i] = 1.0f;  // self-loop contributes 1 to degree
}

__global__ void deg_accum_kernel(const int* __restrict__ dst, float* __restrict__ deg, int E) {
  int i = blockIdx.x * blockDim.x + threadIdx.x;
  int stride = gridDim.x * blockDim.x;
  for (; i < E; i += stride) atomicAdd(&deg[dst[i]], 1.0f);
}

__global__ void deg_finalize_kernel(float* __restrict__ p, int n) {
  int i = blockIdx.x * blockDim.x + threadIdx.x;
  if (i < n) p[i] = rsqrtf(p[i]);
}

// ---------------- WMMA GEMM: H[N,64] = X[N,K] @ W[K,64] ----------------
// One wave -> 16 rows x 64 cols. W^T staged in LDS as bf16.
template<int K>
__global__ __launch_bounds__(256)
void gemm_wmma_kernel(const float* __restrict__ X, const float* __restrict__ W,
                      float* __restrict__ H, int N, int nTiles) {
  __shared__ __bf16 sWT[HID * K];  // W transposed: [col][k]
  const int tid = threadIdx.x;
  for (int i = tid; i < HID * K; i += 256) {
    int c = i / K, k = i - c * K;
    sWT[i] = (__bf16)W[k * HID + c];
  }
  __syncthreads();

  const int wave = tid >> 5;
  const int lane = tid & 31;
  const int tile = blockIdx.x * 8 + wave;
  if (tile >= nTiles) return;

  const int half = lane >> 4;   // 0: lanes 0-15, 1: lanes 16-31
  const int l15  = lane & 15;
  int row = tile * 16 + l15;
  if (row >= N) row = N - 1;    // clamp (keeps EXEC all-ones for WMMA)
  const int kb = half * 8;      // per 16-bit A-layout: hi lanes start at K=8
  const float* __restrict__ xrow = X + (size_t)row * K;

  v8f acc[4];
  #pragma unroll
  for (int j = 0; j < 4; ++j) acc[j] = (v8f){0.f,0.f,0.f,0.f,0.f,0.f,0.f,0.f};

  for (int k0 = 0; k0 < K; k0 += 32) {
    // A fragment: VGPR0-3 -> K in [kb, kb+8), VGPR4-7 -> K in [kb+16, kb+24)
    v16bf a;
    #pragma unroll
    for (int p = 0; p < 4; ++p) {
      float2 lo = *(const float2*)(xrow + k0 + kb + 2 * p);
      float2 hi = *(const float2*)(xrow + k0 + 16 + kb + 2 * p);
      a[2*p]     = (__bf16)lo.x;  a[2*p+1]   = (__bf16)lo.y;
      a[8+2*p]   = (__bf16)hi.x;  a[8+2*p+1] = (__bf16)hi.y;
    }
    #pragma unroll
    for (int j = 0; j < 4; ++j) {
      const __bf16* wt = sWT + (size_t)(j * 16 + l15) * K;  // column (j*16+l15) of W
      v16bf b;
      #pragma unroll
      for (int p = 0; p < 4; ++p) {
        int k = k0 + kb + 2 * p;
        b[2*p]     = wt[k];       b[2*p+1]   = wt[k + 1];
        b[8+2*p]   = wt[k + 16];  b[8+2*p+1] = wt[k + 17];
      }
      acc[j] = __builtin_amdgcn_wmma_f32_16x16x32_bf16(
          false, a, false, b, (short)0, acc[j], false, false);
    }
  }

  // C/D layout: VGPR i -> M = half*8 + i, N = l15.
  // One base pointer per lane; all stores use constant immediate offsets.
  const int rowBase = tile * 16;
  float* __restrict__ hp = H + (size_t)(rowBase + half * 8) * HID + l15;
  if (rowBase + 16 <= N) {           // wave-uniform fast path (always for N%16==0)
    #pragma unroll
    for (int j = 0; j < 4; ++j) {
      #pragma unroll
      for (int i = 0; i < 8; ++i)
        hp[i * HID + j * 16] = acc[j][i];
    }
  } else {                            // rare ragged tail
    #pragma unroll
    for (int j = 0; j < 4; ++j) {
      #pragma unroll
      for (int i = 0; i < 8; ++i)
        if (rowBase + half * 8 + i < N) hp[i * HID + j * 16] = acc[j][i];
    }
  }
}

// ---------------- edge scatter: agg[dst] += norm * h[src] ----------------
// One wave per edge; lane l handles features (2l, 2l+1). h/agg are L2-resident.
__global__ __launch_bounds__(256)
void scatter_kernel(const float* __restrict__ h, const int* __restrict__ src,
                    const int* __restrict__ dst, const float* __restrict__ dis,
                    float* __restrict__ agg, int E) {
  const int lane = threadIdx.x & 31;
  int w = (blockIdx.x * blockDim.x + threadIdx.x) >> 5;
  const int nw = (gridDim.x * blockDim.x) >> 5;
  for (; w < E; w += nw) {
    int s = src[w], d = dst[w];
    float norm = dis[s] * dis[d];
    float2 v = ((const float2*)(h + (size_t)s * HID))[lane];
    float* ap = agg + (size_t)d * HID + lane * 2;
    atomicAdd(ap,     v.x * norm);
    atomicAdd(ap + 1, v.y * norm);
  }
}

// ---------------- combine: out = relu(agg + h*dis^2 + b) ----------------
__global__ void combine_kernel(const float* __restrict__ agg, const float* __restrict__ h,
                               const float* __restrict__ dis, const float* __restrict__ bias,
                               float* __restrict__ out, int total) {
  int i = blockIdx.x * blockDim.x + threadIdx.x;
  int stride = gridDim.x * blockDim.x;
  for (; i < total; i += stride) {
    int node = i >> 6, c = i & 63;
    float d = dis[node];
    float v = agg[i] + h[i] * d * d + bias[c];
    out[i] = v > 0.f ? v : 0.f;
  }
}

// ---------------- global mean pool ----------------
__global__ void count_kernel(const int* __restrict__ batch, float* __restrict__ cnts, int N) {
  int i = blockIdx.x * blockDim.x + threadIdx.x;
  int stride = gridDim.x * blockDim.x;
  for (; i < N; i += stride) atomicAdd(&cnts[batch[i]], 1.0f);
}

__global__ void pool_kernel(const float* __restrict__ h, const int* __restrict__ batch,
                            float* __restrict__ sums, int total) {
  int i = blockIdx.x * blockDim.x + threadIdx.x;
  int stride = gridDim.x * blockDim.x;
  for (; i < total; i += stride) {
    int node = i >> 6;
    atomicAdd(&sums[batch[node] * HID + (i & 63)], h[i]);
  }
}

// ---------------- tiny MLP head + log_softmax (one block) ----------------
__global__ __launch_bounds__(256)
void mlp_kernel(const float* __restrict__ sums, const float* __restrict__ cnts,
                const float* __restrict__ fW1, const float* __restrict__ fb1,
                const float* __restrict__ fW2, const float* __restrict__ fb2,
                float* __restrict__ out) {
  __shared__ float g[NGRAPH * HID];
  __shared__ float f1[NGRAPH * 32];
  __shared__ float lg[NGRAPH * 16];
  const int tid = threadIdx.x;
  for (int i = tid; i < NGRAPH * HID; i += 256) {
    float c = cnts[i >> 6];
    g[i] = sums[i] / (c > 1.0f ? c : 1.0f);
  }
  __syncthreads();
  for (int o = tid; o < NGRAPH * 32; o += 256) {
    int r = o >> 5, c = o & 31;
    float acc = fb1[c];
    #pragma unroll 8
    for (int k = 0; k < HID; ++k) acc += g[r * HID + k] * fW1[k * 32 + c];
    f1[o] = acc > 0.f ? acc : 0.f;
  }
  __syncthreads();
  for (int o = tid; o < NGRAPH * 16; o += 256) {
    int r = o >> 4, c = o & 15;
    float acc = fb2[c];
    #pragma unroll 8
    for (int k = 0; k < 32; ++k) acc += f1[r * 32 + k] * fW2[k * 16 + c];
    lg[o] = acc;
  }
  __syncthreads();
  if (tid < NGRAPH) {
    float m = -1e30f;
    for (int c = 0; c < 16; ++c) m = fmaxf(m, lg[tid * 16 + c]);
    float s = 0.f;
    for (int c = 0; c < 16; ++c) s += __expf(lg[tid * 16 + c] - m);
    float l = __logf(s);
    for (int c = 0; c < 16; ++c) out[tid * 16 + c] = lg[tid * 16 + c] - m - l;
  }
}

// ---------------- driver ----------------
extern "C" void kernel_launch(void* const* d_in, const int* in_sizes, int n_in,
                              void* d_out, int out_size, void* d_ws, size_t ws_size,
                              hipStream_t stream) {
  const float* x    = (const float*)d_in[0];
  const int*   ei   = (const int*)d_in[1];   // [2, E]
  const int*   batch= (const int*)d_in[2];
  const float* W1   = (const float*)d_in[3];
  const float* b1   = (const float*)d_in[4];
  const float* W2   = (const float*)d_in[5];
  const float* b2   = (const float*)d_in[6];
  const float* W3   = (const float*)d_in[7];
  const float* b3   = (const float*)d_in[8];
  const float* fW1  = (const float*)d_in[9];
  const float* fb1  = (const float*)d_in[10];
  const float* fW2  = (const float*)d_in[11];
  const float* fb2  = (const float*)d_in[12];
  float* out = (float*)d_out;

  const int FIN = 128;
  const int N = in_sizes[0] / FIN;
  const int E = in_sizes[1] / 2;
  const int* src = ei;
  const int* dst = ei + E;

  char* ws = (char*)d_ws;
  auto aln = [](size_t v) { return (v + 255) & ~(size_t)255; };
  size_t off = 0;
  float* dis  = (float*)(ws + off); off += aln((size_t)N * 4);
  float* hA   = (float*)(ws + off); off += aln((size_t)N * HID * 4);
  float* hB   = (float*)(ws + off); off += aln((size_t)N * HID * 4);
  float* hC   = (float*)(ws + off); off += aln((size_t)N * HID * 4);
  float* sums = (float*)(ws + off); off += aln((size_t)NGRAPH * HID * 4);
  float* cnts = (float*)(ws + off); off += aln((size_t)NGRAPH * 4);
  (void)ws_size; (void)n_in; (void)out_size;

  // degree -> dis = deg^{-1/2}
  init_ones_kernel<<<(N + 255) / 256, 256, 0, stream>>>(dis, N);
  deg_accum_kernel<<<1024, 256, 0, stream>>>(dst, dis, E);
  deg_finalize_kernel<<<(N + 255) / 256, 256, 0, stream>>>(dis, N);

  const int nTiles = (N + 15) / 16;
  const int gemmBlocks = (nTiles + 7) / 8;
  const int total = N * HID;

  // layer 1
  gemm_wmma_kernel<128><<<gemmBlocks, 256, 0, stream>>>(x, W1, hA, N, nTiles);
  hipMemsetAsync(hC, 0, (size_t)total * 4, stream);
  scatter_kernel<<<2048, 256, 0, stream>>>(hA, src, dst, dis, hC, E);
  combine_kernel<<<1024, 256, 0, stream>>>(hC, hA, dis, b1, hB, total);
  // layer 2
  gemm_wmma_kernel<64><<<gemmBlocks, 256, 0, stream>>>(hB, W2, hA, N, nTiles);
  hipMemsetAsync(hC, 0, (size_t)total * 4, stream);
  scatter_kernel<<<2048, 256, 0, stream>>>(hA, src, dst, dis, hC, E);
  combine_kernel<<<1024, 256, 0, stream>>>(hC, hA, dis, b2, hB, total);
  // layer 3
  gemm_wmma_kernel<64><<<gemmBlocks, 256, 0, stream>>>(hB, W3, hA, N, nTiles);
  hipMemsetAsync(hC, 0, (size_t)total * 4, stream);
  scatter_kernel<<<2048, 256, 0, stream>>>(hA, src, dst, dis, hC, E);
  combine_kernel<<<1024, 256, 0, stream>>>(hC, hA, dis, b3, hB, total);

  // pool + MLP head
  hipMemsetAsync(sums, 0, (size_t)NGRAPH * HID * 4, stream);
  hipMemsetAsync(cnts, 0, (size_t)NGRAPH * 4, stream);
  count_kernel<<<256, 256, 0, stream>>>(batch, cnts, N);
  pool_kernel<<<1024, 256, 0, stream>>>(hB, batch, sums, total);
  mlp_kernel<<<1, 256, 0, stream>>>(sums, cnts, fW1, fb1, fW2, fb2, out);
}
